// MLPMoE_40939628265544
// MI455X (gfx1250) — matrix operand order repaired and checked by reading, not compiled, not run on hardware
//
#include <hip/hip_runtime.h>
#include <hip/hip_bf16.h>

// ---------------------------------------------------------------------------
// MoE top-2 MLP for MI455X (gfx1250, wave32, WMMA).
//   1) gate: one wave per token -> softmax, top-2, per-expert token lists
//   2) pre-pass: x -> bf16; W1 -> W1^T bf16 [E,H,D]; W2 -> W2^T bf16 [E,D,H]
//      (one-time conversion, so GEMM inner loops are pure b128-load + WMMA)
//   3) GEMM1+GELU (K=D=1024) and GEMM2+scale+atomic-combine (K=H=4096)
//      as bf16 v_wmma_f32_16x16x32_bf16 with f32 accumulation; fragments are
//      direct 16B-aligned global b128 loads per the CDNA5 7.12.2 layouts.
//      Per wave: 2 M-tiles x 4 N-tiles = 8 accumulators; B fragments are
//      reused by two WMMAs each (12 b128 loads : 8 WMMA per K-step).
// Top-2 dispatch = 137 GFLOP (4x less than dense); HBM ~540MB ~= 23us.
// ---------------------------------------------------------------------------

typedef __attribute__((ext_vector_type(16))) __bf16 v16bf;
typedef __attribute__((ext_vector_type(8)))  float  v8f;

union FragAB { v16bf v; uint4 q[2]; };

#define N_TOK 4096
#define DDIM  1024
#define HDIM  4096
#define NEXP  8

__device__ inline float gelu_tanh(float v) {
    const float c = 0.7978845608028654f; // sqrt(2/pi)
    float u = c * (v + 0.044715f * v * v * v);
    return 0.5f * v * (1.0f + tanhf(u));
}

__device__ inline v8f wmma_bf16(const FragAB& a, const FragAB& b, v8f c) {
    return __builtin_amdgcn_wmma_f32_16x16x32_bf16(false, a.v, false, b.v,
                                                   (short)0, c, false, false);
}

// ---------------------------------------------------------------------------
// Kernel: zero output and expert counters.
// ---------------------------------------------------------------------------
__global__ __launch_bounds__(256) void moe_zero(float* __restrict__ out,
                                                int* __restrict__ cnt) {
    size_t i = (size_t)blockIdx.x * 256 + threadIdx.x;
    out[i] = 0.0f;
    if (i < NEXP) cnt[i] = 0;
}

// ---------------------------------------------------------------------------
// Kernel: x fp32 -> bf16 (packed 8B stores).
// ---------------------------------------------------------------------------
__global__ __launch_bounds__(256) void moe_cvt_x(const float* __restrict__ x,
                                                 __bf16* __restrict__ xbf) {
    size_t i = ((size_t)blockIdx.x * 256 + threadIdx.x) * 4;
    float4 v = *(const float4*)(x + i);
    union { __bf16 h[4]; uint2 u; } o;
    o.h[0] = (__bf16)v.x; o.h[1] = (__bf16)v.y;
    o.h[2] = (__bf16)v.z; o.h[3] = (__bf16)v.w;
    *(uint2*)(xbf + i) = o.u;
}

// ---------------------------------------------------------------------------
// Kernel: per-expert tiled transpose + fp32->bf16.
// in: [E][R][C] fp32, out: [E][C][R] bf16. grid = (C/32, R/32, E), 256 thr.
// ---------------------------------------------------------------------------
__global__ __launch_bounds__(256) void moe_transpose_bf16(
    const float* __restrict__ in, __bf16* __restrict__ outp, int R, int C) {
    __shared__ float t[32][33];
    const size_t eoff = (size_t)blockIdx.z * R * C;
    const int cb = blockIdx.x * 32, rb = blockIdx.y * 32;
    {
        int r = threadIdx.x >> 3, c4 = (threadIdx.x & 7) * 4;
        float4 v = *(const float4*)(in + eoff + (size_t)(rb + r) * C + cb + c4);
        t[r][c4 + 0] = v.x; t[r][c4 + 1] = v.y;
        t[r][c4 + 2] = v.z; t[r][c4 + 3] = v.w;
    }
    __syncthreads();
    {
        int c = threadIdx.x >> 3, r4 = (threadIdx.x & 7) * 4;
        union { __bf16 h[4]; uint2 u; } o;
        o.h[0] = (__bf16)t[r4 + 0][c]; o.h[1] = (__bf16)t[r4 + 1][c];
        o.h[2] = (__bf16)t[r4 + 2][c]; o.h[3] = (__bf16)t[r4 + 3][c];
        *(uint2*)(outp + eoff + (size_t)(cb + c) * R + rb + r4) = o.u;
    }
}

// ---------------------------------------------------------------------------
// Kernel: gating. One wave32 per token; lane 0 does softmax + top-2 and
// appends (token, prob) to per-expert lists.
// ---------------------------------------------------------------------------
__global__ __launch_bounds__(256) void moe_gate(const float* __restrict__ x,
                                                const float* __restrict__ gw,
                                                const float* __restrict__ gb,
                                                int* __restrict__ cnt,
                                                int* __restrict__ listTok,
                                                float* __restrict__ listW) {
    int wv = threadIdx.x >> 5, lane = threadIdx.x & 31;
    int n = blockIdx.x * 8 + wv;

    float acc[NEXP];
#pragma unroll
    for (int e = 0; e < NEXP; ++e) acc[e] = 0.0f;

    const float* xr = x + (size_t)n * DDIM;
    for (int t = 0; t < DDIM / 32; ++t) {
        int k = lane + t * 32;
        float xv = xr[k];
        const float* g = gw + (size_t)k * NEXP;
#pragma unroll
        for (int e = 0; e < NEXP; ++e) acc[e] += xv * g[e];
    }
#pragma unroll
    for (int off = 16; off >= 1; off >>= 1) {
#pragma unroll
        for (int e = 0; e < NEXP; ++e) acc[e] += __shfl_xor(acc[e], off, 32);
    }
    if (lane == 0) {
        float l[NEXP], m = -1e30f;
#pragma unroll
        for (int e = 0; e < NEXP; ++e) { l[e] = acc[e] + gb[e]; m = fmaxf(m, l[e]); }
        float p[NEXP], s = 0.0f;
#pragma unroll
        for (int e = 0; e < NEXP; ++e) { p[e] = __expf(l[e] - m); s += p[e]; }
        float inv = 1.0f / s;
        int i0 = 0;
#pragma unroll
        for (int e = 1; e < NEXP; ++e) if (p[e] > p[i0]) i0 = e;
        int i1 = -1;
#pragma unroll
        for (int e = 0; e < NEXP; ++e)
            if (e != i0 && (i1 < 0 || p[e] > p[i1])) i1 = e;

        int pos0 = atomicAdd(&cnt[i0], 1);
        listTok[i0 * N_TOK + pos0] = n;
        listW [i0 * N_TOK + pos0] = p[i0] * inv;
        int pos1 = atomicAdd(&cnt[i1], 1);
        listTok[i1 * N_TOK + pos1] = n;
        listW [i1 * N_TOK + pos1] = p[i1] * inv;
    }
}

__global__ void moe_offsets(const int* __restrict__ cnt, int* __restrict__ offs) {
    if (threadIdx.x == 0) {
        int s = 0;
#pragma unroll
        for (int e = 0; e < NEXP; ++e) { offs[e] = s; s += cnt[e]; }
    }
}

// ---------------------------------------------------------------------------
// GEMM1: h = gelu(x_gathered @ W1[e] + b1[e]) -> bf16 h buffer.
// Block = 32 tokens x 512 H cols; 8 waves; wave = 32x64 slab (8 WMMA accs).
// Invalid rows are pointer-clamped to a valid row (no zero-fill, no branch);
// their garbage accumulators are never stored.
// ---------------------------------------------------------------------------
__global__ __launch_bounds__(256) void moe_gemm1(
    const __bf16* __restrict__ xbf, const __bf16* __restrict__ W1T,
    const float* __restrict__ b1, const int* __restrict__ listTok,
    const int* __restrict__ cnt, const int* __restrict__ offs,
    __bf16* __restrict__ hbuf) {
    const int e = blockIdx.y, tile = blockIdx.x;
    const int count = cnt[e];
    if (tile * 32 >= count) return;

    __shared__ int toks[32];
    const int tid = threadIdx.x;
    if (tid < 32) {
        int i = tile * 32 + tid;
        toks[tid] = (i < count) ? listTok[e * N_TOK + i] : -1;
    }
    __syncthreads();

    const int wv = tid >> 5, lane = tid & 31, lo = lane & 15, hi = lane >> 4;
    const int nbase = blockIdx.z * 512 + wv * 64;

    const int t0 = toks[lo];       // rows 0..15
    const int t1 = toks[16 + lo];  // rows 16..31
    const __bf16* arow0 = xbf + (size_t)(t0 < 0 ? 0 : t0) * DDIM + hi * 8;
    const __bf16* arow1 = xbf + (size_t)(t1 < 0 ? 0 : t1) * DDIM + hi * 8;
    const __bf16* bcol[4];
#pragma unroll
    for (int j = 0; j < 4; ++j)
        bcol[j] = W1T + (size_t)e * HDIM * DDIM +
                  (size_t)(nbase + j * 16 + lo) * DDIM + hi * 8;

    v8f acc0[4] = {{}, {}, {}, {}};
    v8f acc1[4] = {{}, {}, {}, {}};
#pragma unroll 2
    for (int kb = 0; kb < DDIM; kb += 32) {
        FragAB a0, a1;
        a0.q[0] = *(const uint4*)(arow0 + kb);      // K = kb+hi*8 .. +7
        a0.q[1] = *(const uint4*)(arow0 + kb + 16); // K = kb+16+hi*8 .. +7
        a1.q[0] = *(const uint4*)(arow1 + kb);
        a1.q[1] = *(const uint4*)(arow1 + kb + 16);
#pragma unroll
        for (int j = 0; j < 4; ++j) {
            FragAB b;
            b.q[0] = *(const uint4*)(bcol[j] + kb);     // K = kb+hi*8 .. +7
            b.q[1] = *(const uint4*)(bcol[j] + kb + 16); // K = kb+16+hi*8 .. +7
            acc0[j] = wmma_bf16(a0, b, acc0[j]);
            acc1[j] = wmma_bf16(a1, b, acc1[j]);
        }
    }

    const int rowoff = offs[e];
#pragma unroll
    for (int half = 0; half < 2; ++half) {
        const int mbase = half * 16;
#pragma unroll
        for (int r = 0; r < 8; ++r) {
            int mm = mbase + r + hi * 8;
            int i = tile * 32 + mm;
            if (i < count) {
                size_t hrow = (size_t)(rowoff + i) * HDIM;
#pragma unroll
                for (int j = 0; j < 4; ++j) {
                    int n = nbase + j * 16 + lo;
                    float v = (half ? acc1[j][r] : acc0[j][r]) + b1[e * HDIM + n];
                    hbuf[hrow + n] = (__bf16)gelu_tanh(v);
                }
            }
        }
    }
}

// ---------------------------------------------------------------------------
// GEMM2: out += w_token * (h @ W2[e] + b2[e]). K = H = 4096.
// ---------------------------------------------------------------------------
__global__ __launch_bounds__(256) void moe_gemm2(
    const __bf16* __restrict__ hbuf, const __bf16* __restrict__ W2T,
    const float* __restrict__ b2, const int* __restrict__ listTok,
    const float* __restrict__ listW, const int* __restrict__ cnt,
    const int* __restrict__ offs, float* __restrict__ out) {
    const int e = blockIdx.y, tile = blockIdx.x;
    const int count = cnt[e];
    if (tile * 32 >= count) return;

    __shared__ int toks[32];
    __shared__ float wgt[32];
    const int tid = threadIdx.x;
    if (tid < 32) {
        int i = tile * 32 + tid;
        bool ok = i < count;
        toks[tid] = ok ? listTok[e * N_TOK + i] : -1;
        wgt[tid]  = ok ? listW[e * N_TOK + i] : 0.0f;
    }
    __syncthreads();

    const int wv = tid >> 5, lane = tid & 31, lo = lane & 15, hi = lane >> 4;
    const int nbase = blockIdx.z * 512 + wv * 64;
    const int rowbase = offs[e] + tile * 32;

    // clamp invalid rows to a valid row of this expert's segment
    const int r0 = (tile * 32 + lo      < count) ? lo      : 0;
    const int r1 = (tile * 32 + 16 + lo < count) ? 16 + lo : 0;
    const __bf16* arow0 = hbuf + (size_t)(rowbase + r0) * HDIM + hi * 8;
    const __bf16* arow1 = hbuf + (size_t)(rowbase + r1) * HDIM + hi * 8;
    const __bf16* bcol[4];
#pragma unroll
    for (int j = 0; j < 4; ++j)
        bcol[j] = W2T + (size_t)e * DDIM * HDIM +
                  (size_t)(nbase + j * 16 + lo) * HDIM + hi * 8;

    v8f acc0[4] = {{}, {}, {}, {}};
    v8f acc1[4] = {{}, {}, {}, {}};
#pragma unroll 2
    for (int kb = 0; kb < HDIM; kb += 32) {
        FragAB a0, a1;
        a0.q[0] = *(const uint4*)(arow0 + kb);
        a0.q[1] = *(const uint4*)(arow0 + kb + 16);
        a1.q[0] = *(const uint4*)(arow1 + kb);
        a1.q[1] = *(const uint4*)(arow1 + kb + 16);
#pragma unroll
        for (int j = 0; j < 4; ++j) {
            __builtin_prefetch(bcol[j] + kb + 128, 0, 0); // -> global_prefetch_b8
            FragAB b;
            b.q[0] = *(const uint4*)(bcol[j] + kb);
            b.q[1] = *(const uint4*)(bcol[j] + kb + 16);
            acc0[j] = wmma_bf16(a0, b, acc0[j]);
            acc1[j] = wmma_bf16(a1, b, acc1[j]);
        }
    }

#pragma unroll
    for (int half = 0; half < 2; ++half) {
        const int mbase = half * 16;
#pragma unroll
        for (int r = 0; r < 8; ++r) {
            int mm = mbase + r + hi * 8;
            int i = tile * 32 + mm;
            if (i < count) {
                int t = toks[mm];
                float w = wgt[mm];
#pragma unroll
                for (int j = 0; j < 4; ++j) {
                    int n = nbase + j * 16 + lo;
                    float a = half ? acc1[j][r] : acc0[j][r];
                    atomicAdd(&out[(size_t)t * DDIM + n],
                              w * (a + b2[e * DDIM + n]));
                }
            }
        }
    }
}

// ---------------------------------------------------------------------------
// Launch. Workspace layout (needs ~208MB):
//   [0,32)    cnt[8]        [64,96)  offs[8]
//   [128, +128KB)  listTok  ; then 128KB listW
//   [512KB, +8MB)  xbf  [N][D] bf16
//   [16MB,  +64MB) hbuf [2N][H] bf16 (compact per-expert rows)
//   [80MB,  +64MB) W1T  [E][H][D] bf16
//   [144MB, +64MB) W2T  [E][D][H] bf16
// ---------------------------------------------------------------------------
extern "C" void kernel_launch(void* const* d_in, const int* in_sizes, int n_in,
                              void* d_out, int out_size, void* d_ws, size_t ws_size,
                              hipStream_t stream) {
    const float* x  = (const float*)d_in[0];
    const float* gw = (const float*)d_in[1];
    const float* gb = (const float*)d_in[2];
    const float* W1 = (const float*)d_in[3];
    const float* b1 = (const float*)d_in[4];
    const float* W2 = (const float*)d_in[5];
    const float* b2 = (const float*)d_in[6];
    float* out = (float*)d_out;

    char* ws = (char*)d_ws;
    int*    cnt     = (int*)(ws);
    int*    offs    = (int*)(ws + 64);
    int*    listTok = (int*)(ws + 128);
    float*  listW   = (float*)(ws + 128 + NEXP * N_TOK * 4);
    __bf16* xbf     = (__bf16*)(ws + (size_t)(512 * 1024));
    __bf16* hbuf    = (__bf16*)(ws + (size_t)16 * 1024 * 1024);
    __bf16* W1T     = (__bf16*)(ws + (size_t)80 * 1024 * 1024);
    __bf16* W2T     = (__bf16*)(ws + (size_t)144 * 1024 * 1024);

    moe_zero<<<(N_TOK * DDIM) / 256, 256, 0, stream>>>(out, cnt);
    moe_cvt_x<<<(N_TOK * DDIM) / 1024, 256, 0, stream>>>(x, xbf);
    moe_transpose_bf16<<<dim3(HDIM / 32, DDIM / 32, NEXP), 256, 0, stream>>>(
        W1, W1T, DDIM, HDIM);
    moe_transpose_bf16<<<dim3(DDIM / 32, HDIM / 32, NEXP), 256, 0, stream>>>(
        W2, W2T, HDIM, DDIM);
    moe_gate<<<N_TOK / 8, 256, 0, stream>>>(x, gw, gb, cnt, listTok, listW);
    moe_offsets<<<1, 32, 0, stream>>>(cnt, offs);

    dim3 g1(N_TOK / 32, NEXP, HDIM / 512);
    moe_gemm1<<<g1, 256, 0, stream>>>(xbf, W1T, b1, listTok, cnt, offs, hbuf);

    dim3 g2(N_TOK / 32, NEXP, DDIM / 512);
    moe_gemm2<<<g2, 256, 0, stream>>>(hbuf, W2T, b2, listTok, listW, cnt, offs, out);
}
